// Manager_73340861547085
// MI455X (gfx1250) — matrix operand order repaired and checked by reading, not compiled, not run on hardware
//
#include <hip/hip_runtime.h>
#include <hip/hip_bf16.h>

// ---------------------------------------------------------------------------
// Problem sizes (fixed by the reference)
// ---------------------------------------------------------------------------
#define B_SZ   16
#define L_SZ   2048
#define DM     1024          // D_MODEL (K)
#define DG     512           // D_GOAL  (N)
#define M_TOT  (B_SZ * L_SZ) // 32768   (M)

// GEMM tiling
#define BM  128
#define BN  128
#define BK  32
#define LDA 40               // padded K-stride (bf16 elems): 80 B, 16B aligned

// workspace layout (bytes)
#define WS_SUM   0           // double
#define WS_SQ    8           // double
#define WS_NOISE 64          // float[512]
#define WS_IDX   4096        // int[32768]
#define WS_G     (1 << 18)   // float[M_TOT*DG] = 64 MB @ 256 KB offset

typedef __attribute__((ext_vector_type(16))) __bf16 v16bf;
typedef __attribute__((ext_vector_type(8)))  __bf16 v8bf;
typedef __attribute__((ext_vector_type(8)))  float  v8f;

// ---------------------------------------------------------------------------
// 0) zero the global sum / sumsq accumulators (ws is poisoned by the harness)
// ---------------------------------------------------------------------------
__global__ void zero_accum_kernel(double* accum) {
    if (threadIdx.x < 2) accum[threadIdx.x] = 0.0;
}

// ---------------------------------------------------------------------------
// 1) bf16x3 split-precision GEMM:  g = x @ W^T + b, plus sum/sumsq reduction
//    block = 256 threads (8 wave32), computes a 128x128 tile of g.
//    wave grid 4(M) x 2(N); each wave: 2x4 subtiles of 16x16 (v8f accums).
//    Software-pipelined: next K-tile's global loads are issued while the
//    current K-tile's WMMAs run; tile k+2 is prefetched into cache.
// ---------------------------------------------------------------------------
__global__ __launch_bounds__(256) void gemm_bf16x3_kernel(
    const float* __restrict__ x,     // [M_TOT, DM]
    const float* __restrict__ Wm,    // [DG, DM]
    const float* __restrict__ bias,  // [DG]
    float* __restrict__ g,           // [M_TOT, DG]
    double* __restrict__ accum)      // [2] sum, sumsq
{
    __shared__ __attribute__((aligned(16))) __bf16 Ah[BM * LDA];
    __shared__ __attribute__((aligned(16))) __bf16 Al[BM * LDA];
    __shared__ __attribute__((aligned(16))) __bf16 Bh[BN * LDA];
    __shared__ __attribute__((aligned(16))) __bf16 Bl[BN * LDA];

    const int tid   = threadIdx.x;
    const int lane  = tid & 31;
    const int wave  = tid >> 5;
    const int waveM = wave & 3;   // 0..3 -> 32-row slice
    const int waveN = wave >> 2;  // 0..1 -> 64-col slice
    const int mBase = blockIdx.y * BM;
    const int nBase = blockIdx.x * BN;

    v8f acc[2][4] = {};           // zero-init f32 accumulators

    // staging map: thread -> (row, 16-wide col chunk) of the 128x32 f32 tile
    const int lr = tid >> 1;
    const int lc = (tid & 1) * 16;

    const int grp = lane >> 4;    // lane group 0/1 (K-half selector)
    const int lm  = lane & 15;

    const float* xrow = x  + (size_t)(mBase + lr) * DM + lc;
    const float* wrow = Wm + (size_t)(nBase + lr) * DM + lc;

    // ---- prologue: stage K-tile 0 into registers
    float4 stA[4], stB[4];
    #pragma unroll
    for (int e = 0; e < 4; ++e) {
        stA[e] = *(const float4*)(xrow + 4 * e);
        stB[e] = *(const float4*)(wrow + 4 * e);
    }

    for (int k0 = 0; k0 < DM; k0 += BK) {
        // ---- convert staged f32 regs -> (hi,lo) bf16 split into LDS
        #pragma unroll
        for (int e = 0; e < 4; ++e) {
            float av[4] = {stA[e].x, stA[e].y, stA[e].z, stA[e].w};
            float bv[4] = {stB[e].x, stB[e].y, stB[e].z, stB[e].w};
            #pragma unroll
            for (int q = 0; q < 4; ++q) {
                __bf16 ha = (__bf16)av[q];
                Ah[lr * LDA + lc + 4 * e + q] = ha;
                Al[lr * LDA + lc + 4 * e + q] = (__bf16)(av[q] - (float)ha);
                __bf16 hb = (__bf16)bv[q];
                Bh[lr * LDA + lc + 4 * e + q] = hb;
                Bl[lr * LDA + lc + 4 * e + q] = (__bf16)(bv[q] - (float)hb);
            }
        }
        __syncthreads();

        // ---- A fragments (ISA 16-bit A layout: lane grp 0 -> K {0..7,16..23},
        //      grp 1 -> K {8..15,24..31}; m = lane%16 + subtile offset)
        v16bf a_hi[2], a_lo[2];
        #pragma unroll
        for (int i = 0; i < 2; ++i) {
            const int m = waveM * 32 + i * 16 + lm;
            v8bf r0 = *(const v8bf*)&Ah[m * LDA + grp * 8];
            v8bf r1 = *(const v8bf*)&Ah[m * LDA + 16 + grp * 8];
            v8bf s0 = *(const v8bf*)&Al[m * LDA + grp * 8];
            v8bf s1 = *(const v8bf*)&Al[m * LDA + 16 + grp * 8];
            #pragma unroll
            for (int e = 0; e < 8; ++e) {
                a_hi[i][e] = r0[e]; a_hi[i][e + 8] = r1[e];
                a_lo[i][e] = s0[e]; a_lo[i][e + 8] = s1[e];
            }
        }

        // ---- pipeline: issue next K-tile's global loads now (overlap WMMA),
        //      and prefetch the tile after that into cache.
        const int kn = k0 + BK;
        if (kn < DM) {
            #pragma unroll
            for (int e = 0; e < 4; ++e) {
                stA[e] = *(const float4*)(xrow + kn + 4 * e);
                stB[e] = *(const float4*)(wrow + kn + 4 * e);
            }
            if (kn + BK < DM) {
                __builtin_prefetch(xrow + kn + BK, 0, 1);
                __builtin_prefetch(wrow + kn + BK, 0, 1);
            }
        }

        // ---- B fragments (lane -> N = lane%16, K {8*(lane/16)+0..7, 16+8*(lane/16)+0..7}) + MMA
        #pragma unroll
        for (int j = 0; j < 4; ++j) {
            const int n  = waveN * 64 + j * 16 + lm;
            const int kb = grp * 8;
            v8bf r0 = *(const v8bf*)&Bh[n * LDA + kb];
            v8bf r1 = *(const v8bf*)&Bh[n * LDA + kb + 16];
            v8bf s0 = *(const v8bf*)&Bl[n * LDA + kb];
            v8bf s1 = *(const v8bf*)&Bl[n * LDA + kb + 16];
            v16bf b_hi, b_lo;
            #pragma unroll
            for (int e = 0; e < 8; ++e) {
                b_hi[e] = r0[e]; b_hi[e + 8] = r1[e];
                b_lo[e] = s0[e]; b_lo[e + 8] = s1[e];
            }
            #pragma unroll
            for (int i = 0; i < 2; ++i) {
                acc[i][j] = __builtin_amdgcn_wmma_f32_16x16x32_bf16(
                    false, a_hi[i], false, b_hi, (short)0, acc[i][j], false, false);
                acc[i][j] = __builtin_amdgcn_wmma_f32_16x16x32_bf16(
                    false, a_lo[i], false, b_hi, (short)0, acc[i][j], false, false);
                acc[i][j] = __builtin_amdgcn_wmma_f32_16x16x32_bf16(
                    false, a_hi[i], false, b_lo, (short)0, acc[i][j], false, false);
            }
        }
        __syncthreads();   // LDS consumed; safe to overwrite next iteration
    }

    // ---- epilogue: bias add, store g, wave-level sum/sumsq reduction
    float s = 0.f, sq = 0.f;
    #pragma unroll
    for (int i = 0; i < 2; ++i) {
        #pragma unroll
        for (int j = 0; j < 4; ++j) {
            const int row0 = mBase + waveM * 32 + i * 16 + grp * 8;
            const int col  = nBase + waveN * 64 + j * 16 + lm;
            const float bv = bias[col];
            #pragma unroll
            for (int v = 0; v < 8; ++v) {
                float val = acc[i][j][v] + bv;   // C layout: lane grp -> M+8
                g[(size_t)(row0 + v) * DG + col] = val;
                s += val; sq += val * val;
            }
        }
    }
    #pragma unroll
    for (int off = 16; off; off >>= 1) {
        s  += __shfl_xor(s,  off);
        sq += __shfl_xor(sq, off);
    }
    if (lane == 0) {
        atomicAdd(&accum[0], (double)s);
        atomicAdd(&accum[1], (double)sq);
    }
}

// ---------------------------------------------------------------------------
// 2) noise[i] = 0.5*mean/10 + (sqrt(var)/5) * noise_base[i]
// ---------------------------------------------------------------------------
__global__ void noise_kernel(const double* __restrict__ accum,
                             const float* __restrict__ noise_base,
                             float* __restrict__ noise)
{
    const int i = blockIdx.x * blockDim.x + threadIdx.x;
    if (i >= DG) return;
    const double N    = (double)M_TOT * (double)DG;
    const double mean = accum[0] / N;
    double var        = accum[1] / N - mean * mean;
    if (var < 0.0) var = 0.0;
    const float m  = (float)mean * 0.1f;             // /MEAN_FACTOR
    const float sd = sqrtf((float)var) * 0.2f;       // /STD_FACTOR
    noise[i] = 0.5f * m + sd * noise_base[i];
}

// ---------------------------------------------------------------------------
// 3) segmented forward-fill indices: one wave32 per batch row, shfl max-scan
// ---------------------------------------------------------------------------
__global__ void idx_kernel(const int* __restrict__ mask, int* __restrict__ idx)
{
    const int b = blockIdx.x;
    const int lane = threadIdx.x;
    int running = 0;
    for (int t0 = 0; t0 < L_SZ; t0 += 32) {
        const int t = t0 + lane;
        const int flag = (t == 0) ? 1 : (mask[b * L_SZ + t - 1] != 0);
        int val = flag ? t : 0;
        #pragma unroll
        for (int off = 1; off < 32; off <<= 1) {
            int up = __shfl_up(val, off);
            if (lane >= off) val = max(val, up);
        }
        val = max(val, running);
        idx[b * L_SZ + t] = val;
        running = __shfl(val, 31);
    }
}

// ---------------------------------------------------------------------------
// 4) out[b,t,:] = g[b, idx[b,t], :] + noise[:]   (float4 traffic)
// ---------------------------------------------------------------------------
__global__ __launch_bounds__(128) void gather_kernel(
    const float* __restrict__ g, const int* __restrict__ idx,
    const float* __restrict__ noise, float* __restrict__ out)
{
    const int gid = blockIdx.x;          // b*L + t
    const int b   = gid >> 11;           // /2048
    const int src = idx[gid];
    const float4* grow = (const float4*)(g + ((size_t)b * L_SZ + src) * DG);
    const float4* nz   = (const float4*)noise;
    float4*       orow = (float4*)(out + (size_t)gid * DG);
    const int tid = threadIdx.x;         // 128 threads * float4 = 512 floats
    float4 gv = grow[tid];
    float4 nv = nz[tid];
    orow[tid] = make_float4(gv.x + nv.x, gv.y + nv.y, gv.z + nv.z, gv.w + nv.w);
}

// ---------------------------------------------------------------------------
extern "C" void kernel_launch(void* const* d_in, const int* in_sizes, int n_in,
                              void* d_out, int out_size, void* d_ws, size_t ws_size,
                              hipStream_t stream)
{
    const float* x     = (const float*)d_in[0];  // [16,2048,1024]
    const int*   mask  = (const int*)  d_in[1];  // [16,2048] bool->int
    const float* Wm    = (const float*)d_in[2];  // [512,1024]
    const float* bias  = (const float*)d_in[3];  // [512]
    const float* nbase = (const float*)d_in[4];  // [512]
    float* out = (float*)d_out;                  // [16,2048,512]

    char* ws      = (char*)d_ws;
    double* accum = (double*)(ws + WS_SUM);
    float*  noise = (float*) (ws + WS_NOISE);
    int*    idx   = (int*)   (ws + WS_IDX);
    float*  g     = (float*) (ws + WS_G);

    zero_accum_kernel<<<1, 32, 0, stream>>>(accum);
    gemm_bf16x3_kernel<<<dim3(DG / BN, M_TOT / BM), 256, 0, stream>>>(
        x, Wm, bias, g, accum);
    noise_kernel<<<(DG + 255) / 256, 256, 0, stream>>>(accum, nbase, noise);
    idx_kernel<<<B_SZ, 32, 0, stream>>>(mask, idx);
    gather_kernel<<<M_TOT, 128, 0, stream>>>(g, idx, noise, out);
}
